// CoreAttention_75333726372169
// MI455X (gfx1250) — matrix operand order repaired and checked by reading, not compiled, not run on hardware
//
#include <hip/hip_runtime.h>

typedef _Float16 v16h __attribute__((ext_vector_type(16)));
typedef _Float16 v8h  __attribute__((ext_vector_type(8)));
typedef _Float16 v4h  __attribute__((ext_vector_type(4)));
typedef float    v8f  __attribute__((ext_vector_type(8)));
typedef int      i32x4 __attribute__((vector_size(16)));

#define SEQ_LEN  4096
#define N_HEADS  16
#define HEAD_DIM 128
#define BLOCK_Q  128           // q rows per block (8 waves x 16)
#define KT       32            // key tile
#define NWAVES   8
#define SCALE    0.08838834764831845f   // 1/sqrt(128)
#define ZETA     1.0f
#define GAMMA    (-0.03f)
#define NEG_BIG  (-1.0e30f)

// ---- CDNA5 async global->LDS path (guarded; falls back to sync copies) ----
#if defined(__has_builtin)
# if __has_builtin(__builtin_amdgcn_global_load_async_to_lds_b128)
#  define HAVE_ASYNC 1
# endif
# if __has_builtin(__builtin_amdgcn_s_wait_asynccnt)
#  define HAVE_WAIT_ASYNC 1
# endif
#endif
#ifndef HAVE_ASYNC
# define HAVE_ASYNC 0
#endif
#ifndef HAVE_WAIT_ASYNC
# define HAVE_WAIT_ASYNC 0
#endif

__device__ __forceinline__ v8f wmma_f16(v16h a, v16h b, v8f c) {
  // D = A(16x32 f16) * B(32x16 f16) + C(16x16 f32)
  return __builtin_amdgcn_wmma_f32_16x16x32_f16(false, a, false, b, (short)0, c,
                                                false, false);
}

__device__ __forceinline__ void wait_async() {
#if HAVE_ASYNC
# if HAVE_WAIT_ASYNC
  __builtin_amdgcn_s_wait_asynccnt(0);
# else
  asm volatile("s_wait_asynccnt 0x0" ::: "memory");
# endif
#endif
}

// Issue one KTxHEAD_DIM f32 tile: global -> LDS raw buffer (async if available).
__device__ __forceinline__ void issue_tile(const float* __restrict__ G, int kbase,
                                           int h, float* dst, int tid) {
#pragma unroll
  for (int it = 0; it < 4; ++it) {
    const int f   = tid + 256 * it;     // float4 slot 0..1023
    const int key = f >> 5;             // 0..31
    const int d4  = (f & 31) * 4;       // 0..124
    const float* src = G + ((size_t)((kbase + key) * N_HEADS + h)) * HEAD_DIM + d4;
    float* d = dst + key * HEAD_DIM + d4;
#if HAVE_ASYNC
    __builtin_amdgcn_global_load_async_to_lds_b128(
        (__attribute__((address_space(1))) i32x4*)src,
        (__attribute__((address_space(3))) i32x4*)d, 0, 0);
#else
    *(float4*)d = *(const float4*)src;
#endif
  }
}

// LDS raw f32 -> transposed f16 Kt[d][key] (scatter stores, can't pack)
__device__ __forceinline__ void convert_K(const float* raw, _Float16 (*Kt)[KT], int tid) {
#pragma unroll
  for (int it = 0; it < 4; ++it) {
    const int f   = tid + 256 * it;
    const int key = f >> 5;
    const int d4  = (f & 31) * 4;
    float4 g = *(const float4*)(raw + key * HEAD_DIM + d4);
    Kt[d4 + 0][key] = (_Float16)g.x;
    Kt[d4 + 1][key] = (_Float16)g.y;
    Kt[d4 + 2][key] = (_Float16)g.z;
    Kt[d4 + 3][key] = (_Float16)g.w;
  }
}

// LDS raw f32 -> row-major f16 Vt[key][d] (packed 8B ds_store_b64)
__device__ __forceinline__ void convert_V(const float* raw, _Float16 (*Vt)[HEAD_DIM], int tid) {
#pragma unroll
  for (int it = 0; it < 4; ++it) {
    const int f   = tid + 256 * it;
    const int key = f >> 5;
    const int d4  = (f & 31) * 4;
    float4 g = *(const float4*)(raw + key * HEAD_DIM + d4);
    v4h p = {(_Float16)g.x, (_Float16)g.y, (_Float16)g.z, (_Float16)g.w};
    *(v4h*)(&Vt[key][d4]) = p;
  }
}

__global__ __launch_bounds__(256, 1)
void flash_attn_clipped(const float* __restrict__ Q, const float* __restrict__ K,
                        const float* __restrict__ V, float* __restrict__ O) {
  // Double-buffered raw f32 tiles (async landing zone) + f16 compute tiles.
  __shared__ __align__(64) float    Kraw[2][KT][HEAD_DIM];   // 32 KB
  __shared__ __align__(64) float    Vraw[2][KT][HEAD_DIM];   // 32 KB
  __shared__ __align__(64) _Float16 Kt[HEAD_DIM][KT];        // 8 KB  [d][key]
  __shared__ __align__(64) _Float16 Vt[KT][HEAD_DIM];        // 8 KB  [key][d]
  __shared__ __align__(64) _Float16 Pb[NWAVES][16][KT];      // 8 KB

  const int tid  = threadIdx.x;
  const int lane = tid & 31;
  const int wave = tid >> 5;
  const int hi   = lane >> 4;     // wave half
  const int nlow = lane & 15;     // A-row / C-column index for this lane

  const int h    = blockIdx.y;
  const int q0   = blockIdx.x * BLOCK_Q;
  const int qw   = q0 + wave * 16;        // wave's first q row
  const int qrow = qw + nlow;             // this lane's A-matrix row (Q row)

  // ---------- load Q A-fragments directly from global (scaled, f32->f16) ----
  // 16-bit A 16x32 layout: elements 0..7 cover d = ds*32 + 8*hi + [0..7],
  // elements 8..15 cover d = ds*32 + 16 + 8*hi + [0..7].
  v16h qfrag[4];
  {
    const float* qb = Q + ((size_t)(qrow * N_HEADS + h)) * HEAD_DIM;
#pragma unroll
    for (int ds = 0; ds < 4; ++ds) {
      const int d0 = ds * 32 + 8 * hi;
      float4 f0 = *(const float4*)(qb + d0);
      float4 f1 = *(const float4*)(qb + d0 + 4);
      float4 f2 = *(const float4*)(qb + d0 + 16);
      float4 f3 = *(const float4*)(qb + d0 + 20);
      float tf[16] = {f0.x, f0.y, f0.z, f0.w, f1.x, f1.y, f1.z, f1.w,
                      f2.x, f2.y, f2.z, f2.w, f3.x, f3.y, f3.z, f3.w};
      v16h a;
#pragma unroll
      for (int i = 0; i < 16; ++i) a[i] = (_Float16)(tf[i] * SCALE);
      qfrag[ds] = a;
    }
  }

  const int ntiles = (q0 + BLOCK_Q) / KT;   // causal: keys <= q0+127
  const int rowb   = qw + 8 * hi;           // lane's first C/D row

  // =============== PASS 1: row max m and denominator l ======================
  float mrun[8], lrun[8];
#pragma unroll
  for (int r = 0; r < 8; ++r) { mrun[r] = NEG_BIG; lrun[r] = 0.0f; }

  issue_tile(K, 0, h, &Kraw[0][0][0], tid);   // prologue: tile 0 in flight
  for (int kt = 0; kt < ntiles; ++kt) {
    const int cur   = kt & 1;
    const int kbase = kt * KT;
    wait_async();            // this wave's async writes have landed
    __syncthreads();         // everyone's have landed; prev readers done
    if (kt + 1 < ntiles)
      issue_tile(K, kbase + KT, h, &Kraw[cur ^ 1][0][0], tid);
    convert_K(&Kraw[cur][0][0], Kt, tid);
    __syncthreads();
    if (kbase > qw + 15) continue;        // fully masked for this wave

    // two 16x16 score tiles: B column (key) == nlow in Kt, elements over d
    v8f acc0 = {}, acc1 = {};
#pragma unroll
    for (int ds = 0; ds < 4; ++ds) {
      v16h b0, b1;
#pragma unroll
      for (int i = 0; i < 16; ++i) {
        const int d = ds * 32 + (i < 8 ? 8 * hi + i : 16 + 8 * hi + (i - 8));
        b0[i] = Kt[d][nlow];
        b1[i] = Kt[d][16 + nlow];
      }
      acc0 = wmma_f16(qfrag[ds], b0, acc0);
      acc1 = wmma_f16(qfrag[ds], b1, acc1);
    }
    float s0[8], s1[8];
    if (kbase + KT - 1 > qw) {           // partial tile: apply causal mask
      const int k0g = kbase + nlow, k1g = k0g + 16;
#pragma unroll
      for (int r = 0; r < 8; ++r) {
        s0[r] = (k0g > rowb + r) ? NEG_BIG : acc0[r];
        s1[r] = (k1g > rowb + r) ? NEG_BIG : acc1[r];
      }
    } else {                             // interior tile: no mask VALU
#pragma unroll
      for (int r = 0; r < 8; ++r) { s0[r] = acc0[r]; s1[r] = acc1[r]; }
    }
#pragma unroll
    for (int r = 0; r < 8; ++r) {
      float v = fmaxf(s0[r], s1[r]);
#pragma unroll
      for (int off = 1; off < 16; off <<= 1) v = fmaxf(v, __shfl_xor(v, off, 32));
      const float mn = fmaxf(mrun[r], v);
      float e = __expf(s0[r] - mn) + __expf(s1[r] - mn);
#pragma unroll
      for (int off = 1; off < 16; off <<= 1) e += __shfl_xor(e, off, 32);
      lrun[r] = lrun[r] * __expf(mrun[r] - mn) + e;
      mrun[r] = mn;
    }
  }

  float invl[8];
#pragma unroll
  for (int r = 0; r < 8; ++r) invl[r] = 1.0f / lrun[r];

  // =============== PASS 2: p = clip((Z-G)*softmax + G), O += P*V ============
  v8f oacc[8] = {};

  __syncthreads();   // pass-1 raw-buffer readers done before re-issuing buf 0
  issue_tile(K, 0, h, &Kraw[0][0][0], tid);
  issue_tile(V, 0, h, &Vraw[0][0][0], tid);
  for (int kt = 0; kt < ntiles; ++kt) {
    const int cur   = kt & 1;
    const int kbase = kt * KT;
    wait_async();
    __syncthreads();
    if (kt + 1 < ntiles) {
      issue_tile(K, kbase + KT, h, &Kraw[cur ^ 1][0][0], tid);
      issue_tile(V, kbase + KT, h, &Vraw[cur ^ 1][0][0], tid);
    }
    convert_K(&Kraw[cur][0][0], Kt, tid);
    convert_V(&Vraw[cur][0][0], Vt, tid);
    __syncthreads();
    if (kbase > qw + 15) continue;

    v8f acc0 = {}, acc1 = {};
#pragma unroll
    for (int ds = 0; ds < 4; ++ds) {
      v16h b0, b1;
#pragma unroll
      for (int i = 0; i < 16; ++i) {
        const int d = ds * 32 + (i < 8 ? 8 * hi + i : 16 + 8 * hi + (i - 8));
        b0[i] = Kt[d][nlow];
        b1[i] = Kt[d][16 + nlow];
      }
      acc0 = wmma_f16(qfrag[ds], b0, acc0);
      acc1 = wmma_f16(qfrag[ds], b1, acc1);
    }
    float s0[8], s1[8];
    if (kbase + KT - 1 > qw) {           // partial tile: apply causal mask
      const int k0g = kbase + nlow, k1g = k0g + 16;
#pragma unroll
      for (int r = 0; r < 8; ++r) {
        s0[r] = (k0g > rowb + r) ? NEG_BIG : acc0[r];
        s1[r] = (k1g > rowb + r) ? NEG_BIG : acc1[r];
      }
    } else {
#pragma unroll
      for (int r = 0; r < 8; ++r) { s0[r] = acc0[r]; s1[r] = acc1[r]; }
    }
#pragma unroll
    for (int r = 0; r < 8; ++r) {
      float p0 = __expf(s0[r] - mrun[r]) * invl[r];
      float p1 = __expf(s1[r] - mrun[r]) * invl[r];
      p0 = fminf(fmaxf((ZETA - GAMMA) * p0 + GAMMA, 0.0f), 1.0f);
      p1 = fminf(fmaxf((ZETA - GAMMA) * p1 + GAMMA, 0.0f), 1.0f);
      Pb[wave][8 * hi + r][nlow]      = (_Float16)p0;   // C/D layout: col = nlow
      Pb[wave][8 * hi + r][16 + nlow] = (_Float16)p1;
    }
    // LDS store->load within the wave: drain DScnt before re-reading as A-frag
    asm volatile("s_wait_dscnt 0x0" ::: "memory");
    v16h pa;
    {
      v8h lo = *(const v8h*)(&Pb[wave][nlow][8 * hi]);
      v8h hg = *(const v8h*)(&Pb[wave][nlow][16 + 8 * hi]);
#pragma unroll
      for (int i = 0; i < 8; ++i) { pa[i] = lo[i]; pa[8 + i] = hg[i]; }
    }
#pragma unroll
    for (int dn = 0; dn < 8; ++dn) {
      v16h bv;
#pragma unroll
      for (int i = 0; i < 16; ++i)
        bv[i] = Vt[i < 8 ? 8 * hi + i : 16 + 8 * hi + (i - 8)][dn * 16 + nlow];
      oacc[dn] = wmma_f16(pa, bv, oacc[dn]);
    }
  }

  // ---------- write O (C/D layout: row = 8*hi + r, col d = dn*16 + nlow) ----
#pragma unroll
  for (int dn = 0; dn < 8; ++dn) {
#pragma unroll
    for (int r = 0; r < 8; ++r) {
      const int row = qw + 8 * hi + r;
      O[((size_t)(row * N_HEADS + h)) * HEAD_DIM + dn * 16 + nlow] = oacc[dn][r];
    }
  }
}

extern "C" void kernel_launch(void* const* d_in, const int* in_sizes, int n_in,
                              void* d_out, int out_size, void* d_ws, size_t ws_size,
                              hipStream_t stream) {
  (void)in_sizes; (void)n_in; (void)d_ws; (void)ws_size; (void)out_size;
  const float* q = (const float*)d_in[0];
  const float* k = (const float*)d_in[1];
  const float* v = (const float*)d_in[2];
  float* out = (float*)d_out;
  dim3 grid(SEQ_LEN / BLOCK_Q, N_HEADS);
  dim3 block(256);
  flash_attn_clipped<<<grid, block, 0, stream>>>(q, k, v, out);
}
